// Crossattention_19653770347086
// MI455X (gfx1250) — compile-verified, matrix operand order checked by reading, not run on hardware
//
#include <hip/hip_runtime.h>
#include <hip/hip_bf16.h>

typedef _Float16 half_t;
typedef __attribute__((ext_vector_type(16))) _Float16 v16h;
typedef __attribute__((ext_vector_type(8)))  _Float16 v8h;
typedef __attribute__((ext_vector_type(4)))  _Float16 v4h;
typedef __attribute__((ext_vector_type(8)))  float    v8f;
typedef __attribute__((ext_vector_type(4)))  float    v4f;

#define BB 4
#define Cc 64
#define HH 180
#define WW 320
#define AA 32
#define HW (HH*WW)      // 57600
#define NROWS (BB*HH)   // 720
#define TP 128          // pixels per block in 1x1-conv kernels
#define VLD 328         // padded ld for transposed V in LDS (multiple of 8)

// ---------------------------------------------------------------------------
// CDNA5 wave32 16-bit fragment loader (ISA 7.12.2).
// A 16x32: lane row m=lane&15, half hi=lane>>4; VGPRs 0-3 hold K=hi*8..+7
// (8 CONTIGUOUS halves -> one ds_load_b128), VGPRs 4-7 hold K=16+hi*8..+7.
// B 32x16 is symmetric (N on lanes, K in regs): store B transposed
// (Bt[n][k], K contiguous) and the SAME loader builds B fragments.
// ---------------------------------------------------------------------------
__device__ __forceinline__ v16h load_frag(const half_t* __restrict__ src, int ld) {
  int lane = threadIdx.x & 31;
  int row = lane & 15, hi = lane >> 4;
  const half_t* p = src + row * ld + hi * 8;
  v8h lo = *(const v8h*)(p);        // K = hi*8 .. hi*8+7
  v8h up = *(const v8h*)(p + 16);   // K = 16+hi*8 .. +7
  return __builtin_shufflevector(lo, up, 0,1,2,3,4,5,6,7,8,9,10,11,12,13,14,15);
}

__device__ __forceinline__ v8f wmma_f16(v16h a, v16h b, v8f c) {
  return __builtin_amdgcn_wmma_f32_16x16x32_f16(false, a, false, b, (short)0, c,
                                                false, false);
}

// Async global -> LDS copy, 16B per lane.  LDS dest address = low 32 bits of
// the generic shared pointer; tracked by ASYNCcnt.
__device__ __forceinline__ void async_copy_b128(void* lds_dst, const void* gsrc) {
  unsigned int  d = (unsigned int)(unsigned long long)lds_dst;
  unsigned long long g = (unsigned long long)gsrc;
  asm volatile("global_load_async_to_lds_b128 %0, %1, off"
               :: "v"(d), "v"(g) : "memory");
}

// ---------------------------------------------------------------------------
// Kernel 1: vl = xl0 @ feaL^T, vr = xr0 @ feaR^T   -> rows layout [p][c] f16
// ---------------------------------------------------------------------------
__global__ void __launch_bounds__(256)
prep_v_kernel(const float* __restrict__ xl0, const float* __restrict__ xr0,
              const float* __restrict__ wL, const float* __restrict__ wR,
              half_t* __restrict__ vL, half_t* __restrict__ vR) {
  __shared__ __align__(16) half_t xs[2][TP][Cc];   // 32KB  (A: K=c contiguous)
  __shared__ __align__(16) half_t wh[2][Cc][Cc];   // 16KB  (Bt: [o][c], as-is)
  int tid = threadIdx.x;
  int p0  = blockIdx.x * TP;
  int b   = p0 / HW;
  int q0  = p0 - b * HW;

  for (int idx = tid; idx < Cc*Cc; idx += 256) {
    wh[0][0][idx] = (half_t)wL[idx];      // direct copy: Bt[o][c] = W[o][c]
    wh[1][0][idx] = (half_t)wR[idx];
  }
  for (int idx = tid; idx < (TP*Cc)/4; idx += 256) {
    int c = idx >> 5, p4 = (idx & 31) << 2;
    size_t g = (size_t)(b*Cc + c) * HW + q0 + p4;
    v4f a  = *(const v4f*)(xl0 + g);
    v4f bv = *(const v4f*)(xr0 + g);
#pragma unroll
    for (int e = 0; e < 4; ++e) {
      xs[0][p4+e][c] = (half_t)a[e];
      xs[1][p4+e][c] = (half_t)bv[e];
    }
  }
  __syncthreads();

  int wv = tid >> 5, lane = tid & 31;
  int nL = lane & 15, hi = lane >> 4;
  int mt = wv;                                 // 8 waves, 8 M-tiles
  v8f zf = {0.f,0.f,0.f,0.f,0.f,0.f,0.f,0.f};
  for (int side = 0; side < 2; ++side) {
    v16h aLo = load_frag(&xs[side][mt*16][0], Cc);
    v16h aHi = load_frag(&xs[side][mt*16][32], Cc);
    half_t* out = side ? vR : vL;
#pragma unroll
    for (int nt = 0; nt < 4; ++nt) {
      v8f acc = wmma_f16(aLo, load_frag(&wh[side][nt*16][0], Cc), zf);
      acc     = wmma_f16(aHi, load_frag(&wh[side][nt*16][32], Cc), acc);
      int col = nt*16 + nL;
#pragma unroll
      for (int r = 0; r < 8; ++r) {
        int m = r + (hi << 3);
        out[(size_t)(p0 + mt*16 + m) * Cc + col] = (half_t)acc[r];
      }
    }
  }
}

// ---------------------------------------------------------------------------
// Kernel 2: transition (dw3x3 + pw1x1 + residual) -> LayerNorm -> C->A proj
// One block per 64-pixel strip at fixed (b,h).  q out rows layout [p][a] f16.
// ---------------------------------------------------------------------------
__global__ void __launch_bounds__(256)
prep_q_kernel(const float* __restrict__ x,
              const float* __restrict__ dw_w, const float* __restrict__ dw_b,
              const float* __restrict__ pw_w, const float* __restrict__ pw_b,
              const float* __restrict__ ln_w, const float* __restrict__ ln_b,
              const float* __restrict__ to_w, half_t* __restrict__ qout) {
  extern __shared__ char smem[];
  float*  xs   = (float*) (smem);          // [3][66][64] f32  50688B
  half_t* yS   = (half_t*)(smem + 50688);  // [64][64]  f16     8192B (A)
  half_t* pwH  = (half_t*)(smem + 58880);  // [64][64]  f16     8192B (Bt, as-is)
  half_t* toH  = (half_t*)(smem + 67072);  // [32][64]  f16     4096B (Bt, as-is)
  float*  tS   = (float*) (smem + 71168);  // [64][64]  f32    16384B
  half_t* lnS  = (half_t*)(smem + 87552);  // [64][64]  f16     8192B (A)
  float*  dwT  = (float*) (smem + 95744);  // [9][64]           2304B (transposed)
  float*  dwB  = (float*) (smem + 98048);
  float*  pwB  = (float*) (smem + 98304);
  float*  lnwS = (float*) (smem + 98560);
  float*  lnbS = (float*) (smem + 98816);  // total 99072B

  int tid = threadIdx.x;
  int w0  = blockIdx.x * 64;
  int h   = blockIdx.y;
  int b   = blockIdx.z;

  for (int idx = tid; idx < 64*9; idx += 256) {
    int c = idx / 9, k = idx - c*9;
    dwT[k*64 + c] = dw_w[idx];             // dw_w flat [c][9] -> dwT[k][c]
  }
  for (int idx = tid; idx < 64;   idx += 256) {
    dwB[idx] = dw_b[idx]; pwB[idx] = pw_b[idx];
    lnwS[idx] = ln_w[idx]; lnbS[idx] = ln_b[idx];
  }
  for (int idx = tid; idx < 64*64; idx += 256) pwH[idx] = (half_t)pw_w[idx];
  for (int idx = tid; idx < 32*64; idx += 256) toH[idx] = (half_t)to_w[idx];

  // halo-load x rows h-1..h+1, cols w0-1..w0+64, all 64 channels (SAME pad)
  for (int idx = tid; idx < 3*66*64; idx += 256) {
    int wi = idx % 66;
    int rest = idx / 66;
    int c = rest & 63, dr = rest >> 6;
    int gw = w0 - 1 + wi, gh = h - 1 + dr;
    float v = 0.f;
    if (gw >= 0 && gw < WW && gh >= 0 && gh < HH)
      v = x[((size_t)(b*Cc + c) * HH + gh) * WW + gw];
    xs[(dr*66 + wi)*64 + c] = v;
  }
  __syncthreads();

  // depthwise 3x3 + bias -> yS f16 (4 channels per step, b128 LDS loads, b64 store)
  for (int idx = tid; idx < 64*16; idx += 256) {
    int c4 = (idx & 15) << 2;   // channel group of 4
    int wi = idx >> 4;
    v4f acc = *(const v4f*)(dwB + c4);
#pragma unroll
    for (int dr = 0; dr < 3; ++dr)
#pragma unroll
      for (int dj = 0; dj < 3; ++dj) {
        v4f wv = *(const v4f*)(dwT + (dr*3 + dj)*64 + c4);
        v4f xv = *(const v4f*)(xs + (dr*66 + wi + dj)*64 + c4);
        acc += wv * xv;
      }
    v4h hv = { (half_t)acc[0], (half_t)acc[1], (half_t)acc[2], (half_t)acc[3] };
    *(v4h*)(yS + wi*64 + c4) = hv;
  }
  __syncthreads();

  int wv = tid >> 5, lane = tid & 31;
  int nL = lane & 15, hi = lane >> 4;
  v8f zf = {0.f,0.f,0.f,0.f,0.f,0.f,0.f,0.f};

  // pointwise 1x1 via WMMA, + bias + residual -> tS f32
  for (int j = wv; j < 16; j += 8) {
    int mt = j >> 2, nt = j & 3;
    v16h aLo = load_frag(yS + mt*16*64, 64);
    v16h aHi = load_frag(yS + mt*16*64 + 32, 64);
    v8f acc = wmma_f16(aLo, load_frag(pwH + nt*16*64, 64), zf);
    acc     = wmma_f16(aHi, load_frag(pwH + nt*16*64 + 32, 64), acc);
    int o = nt*16 + nL;
#pragma unroll
    for (int r = 0; r < 8; ++r) {
      int wi = mt*16 + r + (hi << 3);
      tS[wi*64 + o] = acc[r] + pwB[o] + xs[(66 + wi + 1)*64 + o];
    }
  }
  __syncthreads();

  // channel LayerNorm per pixel -> lnS f16 (4 threads/pixel, width-4 shuffles)
  {
    int wi = tid >> 2, part = tid & 3;
    const float* trow = tS + wi*64 + part*16;
    float s = 0.f;
#pragma unroll
    for (int o = 0; o < 16; ++o) s += trow[o];
    s += __shfl_xor(s, 1, 4);
    s += __shfl_xor(s, 2, 4);
    float mu = s * (1.f/64.f);
    float vv = 0.f;
#pragma unroll
    for (int o = 0; o < 16; ++o) { float d = trow[o] - mu; vv += d*d; }
    vv += __shfl_xor(vv, 1, 4);
    vv += __shfl_xor(vv, 2, 4);
    float rs = rsqrtf(vv * (1.f/64.f) + 1e-6f);
#pragma unroll
    for (int o = 0; o < 16; ++o) {
      int oc = part*16 + o;
      lnS[wi*64 + oc] = (half_t)(lnwS[oc] * ((trow[o] - mu) * rs) + lnbS[oc]);
    }
  }
  __syncthreads();

  // projection C->A via WMMA (8 tile jobs, one per wave)
  {
    int mt = wv >> 1, nt = wv & 1;
    v16h aLo = load_frag(lnS + mt*16*64, 64);
    v16h aHi = load_frag(lnS + mt*16*64 + 32, 64);
    v8f acc = wmma_f16(aLo, load_frag(toH + nt*16*64, 64), zf);
    acc     = wmma_f16(aHi, load_frag(toH + nt*16*64 + 32, 64), acc);
    int n = b*HH + h;
    int col = nt*16 + nL;
#pragma unroll
    for (int r = 0; r < 8; ++r) {
      int m = r + (hi << 3);
      qout[(size_t)(n*WW + w0 + mt*16 + m) * AA + col] = (half_t)acc[r];
    }
  }
}

// ---------------------------------------------------------------------------
// Kernel 3: per-row flash attention.  One block per row n.
// warp[w][c] = softmax_v( q[w]·k[v] ) @ v[v][c]   (W=320, dqk=32, dv=64)
// Direction L: (q,k,v)=(qL,qR,vR);  direction R: (qR,qL,vL).
// q/k staged via GLOBAL_LOAD_ASYNC_TO_LDS_B128 (ASYNCcnt path, no VGPR
// round-trip); V staged transposed (vT[c][pix], ld=328) for b128 B-frags.
// Row sums computed as P @ ones in a 5th WMMA accumulator.
// ---------------------------------------------------------------------------
__global__ void __launch_bounds__(320)
attn_kernel(const half_t* __restrict__ qrows, const half_t* __restrict__ krows,
            const half_t* __restrict__ vrows, half_t* __restrict__ warp) {
  extern __shared__ char smem[];
  half_t* qS = (half_t*)(smem);           // [320][32]  20480B (A)
  half_t* kS = (half_t*)(smem + 20480);   // [320][32]  20480B (Bt for S)
  half_t* vT = (half_t*)(smem + 40960);   // [64][VLD]  41984B (Bt for P@V)
  half_t* pS = (half_t*)(smem + 82944);   // 10 waves * 16*32   10240B (A)

  int tid = threadIdx.x;
  int n   = blockIdx.x;

  // async DMA q and k straight into LDS (16B per lane per issue)
  const int4* qg = (const int4*)(qrows + (size_t)n * WW * AA);
  const int4* kg = (const int4*)(krows + (size_t)n * WW * AA);
  for (int i = tid; i < (WW*AA)/8; i += 320) {
    async_copy_b128((char*)qS + i*16, qg + i);
    async_copy_b128((char*)kS + i*16, kg + i);
  }
  // transpose V while staging: global [pix][c] -> LDS vT[c][pix]
  const v8h* vg = (const v8h*)(vrows + (size_t)n * WW * Cc);
  for (int i = tid; i < (WW*Cc)/8; i += 320) {
    int pix = i >> 3, c0 = (i & 7) * 8;
    v8h val = vg[i];
#pragma unroll
    for (int e = 0; e < 8; ++e) vT[(c0 + e)*VLD + pix] = val[e];
  }
  asm volatile("s_wait_asynccnt 0x0" ::: "memory");
  __syncthreads();

  int wv = tid >> 5, lane = tid & 31;
  int nL = lane & 15, hi = lane >> 4;
  half_t* pW = pS + wv * 16 * 32;                // per-wave P scratch
  v8f zf = {0.f,0.f,0.f,0.f,0.f,0.f,0.f,0.f};
  v16h ones;
#pragma unroll
  for (int i = 0; i < 16; ++i) ones[i] = (half_t)1.f;

  for (int qt = 0; qt < 2; ++qt) {               // 10 waves * 2 tiles = 20
    int m0 = wv*32 + qt*16;
    v16h aQ = load_frag(qS + m0*AA, AA);         // K=32 = full head dim

    float rmax[8];
    v8f acc[4], rsum;
#pragma unroll
    for (int r = 0; r < 8; ++r) rmax[r] = -1e30f;
#pragma unroll
    for (int j = 0; j < 4; ++j) acc[j] = zf;
    rsum = zf;

    for (int ch = 0; ch < 10; ++ch) {            // 32 k-columns per chunk
      const half_t* kBase = kS + ch*32*AA;
      v8f S0 = wmma_f16(aQ, load_frag(kBase,         AA), zf);
      v8f S1 = wmma_f16(aQ, load_frag(kBase + 16*AA, AA), zf);

#pragma unroll
      for (int r = 0; r < 8; ++r) {
        float t = fmaxf(S0[r], S1[r]);
        t = fmaxf(t, __shfl_xor(t, 1, 32));
        t = fmaxf(t, __shfl_xor(t, 2, 32));
        t = fmaxf(t, __shfl_xor(t, 4, 32));
        t = fmaxf(t, __shfl_xor(t, 8, 32));      // stays within 16-lane group
        float nm = fmaxf(rmax[r], t);
        float sc = __expf(rmax[r] - nm);
        float p0 = __expf(S0[r] - nm);
        float p1 = __expf(S1[r] - nm);
        rmax[r] = nm;
        acc[0][r] *= sc; acc[1][r] *= sc; acc[2][r] *= sc; acc[3][r] *= sc;
        rsum[r]   *= sc;
        // D-layout (m=r+8*hi, n=nL) -> row-major P tile 16x32 in LDS
        int m = r + (hi << 3);
        pW[m*32 + nL]      = (half_t)p0;
        pW[m*32 + 16 + nL] = (half_t)p1;
      }
      asm volatile("s_wait_dscnt 0x0" ::: "memory");   // wave-local LDS RAW
      v16h aP = load_frag(pW, 32);
      rsum = wmma_f16(aP, ones, rsum);           // row sums: P @ 1
      const half_t* vBase = vT + ch*32;          // K offset along pix axis
#pragma unroll
      for (int j = 0; j < 4; ++j)
        acc[j] = wmma_f16(aP, load_frag(vBase + j*16*VLD, VLD), acc[j]);
    }

    // normalize and emit warp rows f16
    float inv[8];
#pragma unroll
    for (int r = 0; r < 8; ++r) inv[r] = 1.f / rsum[r];
#pragma unroll
    for (int j = 0; j < 4; ++j) {
      int col = j*16 + nL;
#pragma unroll
      for (int r = 0; r < 8; ++r) {
        int m = r + (hi << 3);
        warp[(size_t)(n*WW + m0 + m) * Cc + col] =
            (half_t)(acc[j][r] * inv[r]);
      }
    }
  }
}

// ---------------------------------------------------------------------------
// Kernel 4: out = warp @ out_w^T + x0   (f32 output, [B,C,H,W])
// D-layout rows r=0..3 / 4..7 are consecutive pixels -> v4f residual+store.
// ---------------------------------------------------------------------------
__global__ void __launch_bounds__(256)
out_conv_kernel(const half_t* __restrict__ warp, const float* __restrict__ x0,
                const float* __restrict__ ow, float* __restrict__ out) {
  __shared__ __align__(16) half_t wS[TP][Cc];   // 16KB (A)
  __shared__ __align__(16) half_t oH[Cc][Cc];   // 8KB  (Bt, as-is)
  int tid = threadIdx.x;
  int p0  = blockIdx.x * TP;
  int b   = p0 / HW;
  int q0  = p0 - b * HW;

  for (int idx = tid; idx < Cc*Cc; idx += 256) oH[0][idx] = (half_t)ow[idx];
  const int4* wg = (const int4*)(warp + (size_t)p0 * Cc);
  int4* wd = (int4*)&wS[0][0];
  for (int i = tid; i < (TP*Cc)/8; i += 256) wd[i] = wg[i];
  __syncthreads();

  int wv = tid >> 5, lane = tid & 31;
  int nL = lane & 15, hi = lane >> 4;
  int mt = wv;
  v8f zf = {0.f,0.f,0.f,0.f,0.f,0.f,0.f,0.f};
  v16h aLo = load_frag(&wS[mt*16][0], Cc);
  v16h aHi = load_frag(&wS[mt*16][32], Cc);
#pragma unroll
  for (int nt = 0; nt < 4; ++nt) {
    v8f acc = wmma_f16(aLo, load_frag(&oH[nt*16][0], Cc), zf);
    acc     = wmma_f16(aHi, load_frag(&oH[nt*16][32], Cc), acc);
    int o = nt*16 + nL;
    size_t base = (size_t)(b*Cc + o) * HW + q0 + mt*16 + (hi << 3);
    v4f xlo = *(const v4f*)(x0 + base);
    v4f xhi = *(const v4f*)(x0 + base + 4);
    v4f y0, y1;
#pragma unroll
    for (int r = 0; r < 4; ++r) { y0[r] = acc[r] + xlo[r]; y1[r] = acc[r+4] + xhi[r]; }
    *(v4f*)(out + base)     = y0;
    *(v4f*)(out + base + 4) = y1;
  }
}

// ---------------------------------------------------------------------------
extern "C" void kernel_launch(void* const* d_in, const int* in_sizes, int n_in,
                              void* d_out, int out_size, void* d_ws, size_t ws_size,
                              hipStream_t stream) {
  const float* xl0   = (const float*)d_in[0];
  const float* xr0   = (const float*)d_in[1];
  const float* feaL  = (const float*)d_in[2];
  const float* feaR  = (const float*)d_in[3];
  const float* to_l  = (const float*)d_in[4];
  const float* to_r  = (const float*)d_in[5];
  const float* dw_w  = (const float*)d_in[6];
  const float* dw_b  = (const float*)d_in[7];
  const float* pw_w  = (const float*)d_in[8];
  const float* pw_b  = (const float*)d_in[9];
  const float* lnLw  = (const float*)d_in[10];
  const float* lnLb  = (const float*)d_in[11];
  const float* lnRw  = (const float*)d_in[12];
  const float* lnRb  = (const float*)d_in[13];
  const float* out_w = (const float*)d_in[14];
  float* out = (float*)d_out;

  // workspace carve (f16 intermediates), 147.5 MB total
  char* ws = (char*)d_ws;
  half_t* qL = (half_t*)(ws + 0);            // 720*320*32*2 = 14745600
  half_t* qR = (half_t*)(ws + 14745600);
  half_t* vL = (half_t*)(ws + 29491200);     // 720*320*64*2 = 29491200
  half_t* vR = (half_t*)(ws + 58982400);
  half_t* wL = (half_t*)(ws + 88473600);
  half_t* wR = (half_t*)(ws + 117964800);

  const int nPixBlocks = (BB*HW) / TP;       // 1800

  prep_v_kernel<<<dim3(nPixBlocks), dim3(256), 0, stream>>>(
      xl0, xr0, feaL, feaR, vL, vR);

  prep_q_kernel<<<dim3(WW/64, HH, BB), dim3(256), 99072, stream>>>(
      xl0, dw_w, dw_b, pw_w, pw_b, lnLw, lnLb, to_l, qL);
  prep_q_kernel<<<dim3(WW/64, HH, BB), dim3(256), 99072, stream>>>(
      xr0, dw_w, dw_b, pw_w, pw_b, lnRw, lnRb, to_r, qR);

  attn_kernel<<<dim3(NROWS), dim3(320), 93184, stream>>>(qL, qR, vR, wL);
  attn_kernel<<<dim3(NROWS), dim3(320), 93184, stream>>>(qR, qL, vL, wR);

  out_conv_kernel<<<dim3(nPixBlocks), dim3(256), 0, stream>>>(
      wL, xl0, out_w, out);
  out_conv_kernel<<<dim3(nPixBlocks), dim3(256), 0, stream>>>(
      wR, xr0, out_w, out + (size_t)BB*Cc*HW);
}
